// TransformerEncoderLayerLite_36386962932051
// MI455X (gfx1250) — compile-verified, matrix operand order checked by reading, not run on hardware
//
#include <hip/hip_runtime.h>

// ---------------------------------------------------------------------------
// TransformerEncoderLayerLite for MI455X (gfx1250, wave32, WMMA bf16,
// async global->LDS staging double-buffered over K)
// D_MODEL=512 NHEAD=8 HEAD_DIM=64 DFF=2048 RANK=64 L=1024 N=8
// ---------------------------------------------------------------------------

#define D_MODEL 512
#define NHEAD 8
#define HEAD_DIM 64
#define DFF 2048
#define RANK 64
#define SEQ_L 1024
#define BATCH_N 8
#define M_ROWS (SEQ_L * BATCH_N) /* 8192 */
#define LN_EPS 1e-5f

typedef __attribute__((ext_vector_type(16))) __bf16 v16bf;
typedef __attribute__((ext_vector_type(8)))  float  v8f;

#if __has_builtin(__builtin_amdgcn_global_load_async_to_lds_b128)
#define ASYNC_LDS 1
typedef int v4i __attribute__((vector_size(16)));
typedef __attribute__((address_space(1))) v4i* g_v4i_p;  // global
typedef __attribute__((address_space(3))) v4i* l_v4i_p;  // LDS
#else
#define ASYNC_LDS 0
#endif

__device__ __forceinline__ void wait_async0() {
#if __has_builtin(__builtin_amdgcn_s_wait_asynccnt)
  __builtin_amdgcn_s_wait_asynccnt(0);
#else
  asm volatile("s_wait_asynccnt 0x0" ::: "memory");
#endif
}

__device__ __forceinline__ unsigned short f2bf(float f) {
  unsigned u = __float_as_uint(f);
  u += 0x7FFFu + ((u >> 16) & 1u);   // round-to-nearest-even
  return (unsigned short)(u >> 16);
}

union ABu { v16bf v; unsigned u[8]; };

// ---------------------------------------------------------------------------
// Generic batched WMMA GEMM:  C[m,n] = scale * sum_k A[m,k] * Bt[n,k] (+bias)
// A : bf16 (as ushort), row-major stride lda, base = aHi*(z>>3)+aLo*(z&7)
// Bt: bf16 (as ushort), row-major stride ldb (B pre-transposed, k contiguous)
// C : f32 or bf16, element (m,n) at cBase + m*ldcR + n*ldcC
// Block = 256 thr (8 waves), tile 128(M)x64(N); wave w -> rows [w*16,w*16+16)
// K staged through LDS (A 8KB + B 4KB, double buffered) via async-to-LDS.
// ---------------------------------------------------------------------------
template <bool OUT_BF16, bool RELU, bool HAS_BIAS>
__global__ __launch_bounds__(256)
void gemm_wmma_bf16(const unsigned short* __restrict__ A,
                    long aHi, long aLo, int lda,
                    const unsigned short* __restrict__ Bt,
                    long bHi, long bLo, int ldb,
                    void* __restrict__ Cp,
                    long cHi, long cLo, int ldcR, int ldcC,
                    const float* __restrict__ bias,
                    float scale, int K) {
  __shared__ __align__(16) unsigned char smem[2 * (8192 + 4096)];

  const int t    = threadIdx.x;
  const int lane = t & 31;
  const int wave = t >> 5;
  const int z    = blockIdx.z;
  const long aBase = aHi * (long)(z >> 3) + aLo * (long)(z & 7);
  const long bBase = bHi * (long)(z >> 3) + bLo * (long)(z & 7);
  const long cBase = cHi * (long)(z >> 3) + cLo * (long)(z & 7);
  const int half = lane >> 4;   // 16-lane half selects K sub-range
  const int nl   = lane & 15;
  const int mBlk = blockIdx.x * 128;
  const int nBlk = blockIdx.y * 64;

  const unsigned short* Ag = A + aBase + (long)mBlk * lda;
  const unsigned short* Bg = Bt + bBase + (long)nBlk * ldb;

  // cooperative staging: 16B chunks; A tile 128x32 = 512 chunks (2/thread),
  // B tile 64x32 = 256 chunks (1/thread). chunk c -> row c>>2, k-off (c&3)*8
  const int caR = t >> 2, caK = (t & 3) * 8;  // A chunk (also +64 rows for 2nd)

  auto stage = [&](int kk, int buf) {
    unsigned char* sA = smem + buf * 12288;
    unsigned char* sB = sA + 8192;
    const unsigned short* ga0 = Ag + (long)caR * lda + kk + caK;
    const unsigned short* ga1 = Ag + (long)(caR + 64) * lda + kk + caK;
    const unsigned short* gb  = Bg + (long)caR * ldb + kk + caK;
#if ASYNC_LDS
    __builtin_amdgcn_global_load_async_to_lds_b128(
        (g_v4i_p)(void*)ga0, (l_v4i_p)(void*)(sA + t * 16), 0, 0);
    __builtin_amdgcn_global_load_async_to_lds_b128(
        (g_v4i_p)(void*)ga1, (l_v4i_p)(void*)(sA + (t + 256) * 16), 0, 0);
    __builtin_amdgcn_global_load_async_to_lds_b128(
        (g_v4i_p)(void*)gb, (l_v4i_p)(void*)(sB + t * 16), 0, 0);
#else
    *(uint4*)(sA + t * 16)         = *(const uint4*)ga0;
    *(uint4*)(sA + (t + 256) * 16) = *(const uint4*)ga1;
    *(uint4*)(sB + t * 16)         = *(const uint4*)gb;
#endif
  };

  v8f zero = {0.f, 0.f, 0.f, 0.f, 0.f, 0.f, 0.f, 0.f};
  v8f acc[4];
  acc[0] = zero; acc[1] = zero; acc[2] = zero; acc[3] = zero;

  const int nk = K >> 5;
  stage(0, 0);
  for (int kt = 0; kt < nk; ++kt) {
    const int buf = kt & 1;
#if ASYNC_LDS
    wait_async0();
#endif
    __syncthreads();
    if (kt + 1 < nk) stage((kt + 1) << 5, buf ^ 1);

    const uint4* As = (const uint4*)(smem + buf * 12288);
    const uint4* Bs = (const uint4*)(smem + buf * 12288 + 8192);

    // A fragment: row r, bytes [half*16 .. +16) and [32+half*16 .. +16)
    ABu a;
    {
      const int r = wave * 16 + nl;
      uint4 c0 = As[r * 4 + half];
      uint4 c1 = As[r * 4 + 2 + half];
      a.u[0] = c0.x; a.u[1] = c0.y; a.u[2] = c0.z; a.u[3] = c0.w;
      a.u[4] = c1.x; a.u[5] = c1.y; a.u[6] = c1.z; a.u[7] = c1.w;
    }
#pragma unroll
    for (int j = 0; j < 4; ++j) {
      const int n = j * 16 + nl;
      uint4 c0 = Bs[n * 4 + half];
      uint4 c1 = Bs[n * 4 + 2 + half];
      ABu b;
      b.u[0] = c0.x; b.u[1] = c0.y; b.u[2] = c0.z; b.u[3] = c0.w;
      b.u[4] = c1.x; b.u[5] = c1.y; b.u[6] = c1.z; b.u[7] = c1.w;
      acc[j] = __builtin_amdgcn_wmma_f32_16x16x32_bf16(
          false, a.v, false, b.v, (short)0, acc[j], false, false);
    }
  }

  // D layout: lane 0-15 -> N=lane, M=r; lane 16-31 -> N=lane-16, M=r+8
  const int m0 = mBlk + wave * 16;
#pragma unroll
  for (int j = 0; j < 4; ++j) {
    const int ncol = nBlk + j * 16 + nl;
    float bv = 0.f;
    if (HAS_BIAS) bv = bias[ncol];
#pragma unroll
    for (int r = 0; r < 8; ++r) {
      float x = acc[j][r] * scale + bv;
      if (RELU) x = fmaxf(x, 0.f);
      const long o = cBase + (long)(m0 + r + half * 8) * ldcR + (long)ncol * ldcC;
      if constexpr (OUT_BF16)
        ((unsigned short*)Cp)[o] = f2bf(x);
      else
        ((float*)Cp)[o] = x;
    }
  }
}

// ---------------------------------------------------------------------------
// fp32 -> bf16 elementwise
// ---------------------------------------------------------------------------
__global__ void cvt_bf16(const float* __restrict__ in,
                         unsigned short* __restrict__ out, long n) {
  long i = (long)blockIdx.x * 256 + threadIdx.x;
  if (i < n) out[i] = f2bf(in[i]);
}

// in: (R,C) fp32 row-major  ->  out: (C,R) bf16 row-major (transpose)
__global__ void transpose_bf16(const float* __restrict__ in,
                               unsigned short* __restrict__ out, int R, int C) {
  long o = (long)blockIdx.x * 256 + threadIdx.x;
  if (o >= (long)R * C) return;
  int c = (int)(o / R);
  int r = (int)(o % R);
  out[o] = f2bf(in[(long)r * C + c]);
}

// Vbf[(s*8+n)*512 + h*64 + d]  ->  VT[((n*8+h)*64 + d)*1024 + s]
__global__ void transpose_v(const unsigned short* __restrict__ Vbf,
                            unsigned short* __restrict__ VT) {
  long idx = (long)blockIdx.x * 256 + threadIdx.x;  // 4,194,304 total
  int s = (int)(idx & 1023);
  int d = (int)((idx >> 10) & 63);
  int h = (int)((idx >> 16) & 7);
  int n = (int)(idx >> 19);
  VT[idx] = Vbf[((long)s * 8 + n) * 512 + h * 64 + d];
}

// ---------------------------------------------------------------------------
// Exact top-k threshold (radix select) + masked softmax, one block per row.
// Writes bf16 probabilities OVER the fp32 score row (block owns the row).
// ---------------------------------------------------------------------------
__global__ __launch_bounds__(256)
void topk_softmax(float* __restrict__ scores, const float* __restrict__ topk_p) {
  __shared__ float    vals[1024];
  __shared__ unsigned ordv[1024];
  __shared__ unsigned hist[256];
  __shared__ float    red[256];
  __shared__ unsigned sh_p, sh_k;

  const int t = threadIdx.x;
  float* row = scores + (long)blockIdx.x * 1024;

  for (int i = t; i < 1024; i += 256) {
    float f = row[i];
    vals[i] = f;
    unsigned u = __float_as_uint(f);
    ordv[i] = (u >> 31) ? ~u : (u | 0x80000000u);  // order-preserving map
  }
  __syncthreads();

  // k = int(1024 * sigmoid(topk))
  float tk = topk_p[0];
  int kkeep = (int)(1024.f * (1.f / (1.f + __expf(-tk))));
  if (kkeep < 1) kkeep = 1;
  if (kkeep > 1024) kkeep = 1024;

  unsigned prefix = 0, pmask = 0, kk = (unsigned)kkeep;
  for (int level = 0; level < 4; ++level) {
    const int shift = 24 - level * 8;
    hist[t] = 0;
    __syncthreads();
    for (int i = t; i < 1024; i += 256) {
      unsigned o = ordv[i];
      if ((o & pmask) == prefix) atomicAdd(&hist[(o >> shift) & 0xFFu], 1u);
    }
    __syncthreads();
    if (t == 0) {
      unsigned k2 = kk;
      int bin;
      for (bin = 255; bin > 0; --bin) {
        if (k2 <= hist[bin]) break;
        k2 -= hist[bin];
      }
      sh_p = prefix | ((unsigned)bin << shift);
      sh_k = k2;
    }
    __syncthreads();
    prefix = sh_p;
    kk = sh_k;
    pmask |= (0xFFu << shift);
    __syncthreads();
  }
  const unsigned thr = prefix;  // ordered-uint of k-th largest; keep ord >= thr

  // row max (the max is always kept)
  float mx = -3.402823466e38f;
  for (int i = t; i < 1024; i += 256) mx = fmaxf(mx, vals[i]);
  red[t] = mx;
  __syncthreads();
  for (int s = 128; s > 0; s >>= 1) {
    if (t < s) red[t] = fmaxf(red[t], red[t + s]);
    __syncthreads();
  }
  mx = red[0];
  __syncthreads();

  float sum = 0.f;
  for (int i = t; i < 1024; i += 256)
    if (ordv[i] >= thr) sum += __expf(vals[i] - mx);
  red[t] = sum;
  __syncthreads();
  for (int s = 128; s > 0; s >>= 1) {
    if (t < s) red[t] += red[t + s];
    __syncthreads();
  }
  const float inv = 1.f / red[0];
  __syncthreads();

  // overlay bf16 probs on this row's storage (row slot = 4KB; probs use 2KB)
  unsigned short* prow = (unsigned short*)row;
  for (int i = t; i < 1024; i += 256) {
    float p = (ordv[i] >= thr) ? __expf(vals[i] - mx) * inv : 0.f;
    prow[i] = f2bf(p);
  }
}

// ---------------------------------------------------------------------------
// y = LayerNorm(ra + rb) * g + be ; optional bf16 copy for downstream WMMA
// one block (256 thr) per 512-wide row
// ---------------------------------------------------------------------------
__global__ __launch_bounds__(256)
void ln_residual(const float* __restrict__ ra, const float* __restrict__ rb,
                 const float* __restrict__ g, const float* __restrict__ be,
                 float* __restrict__ outF, unsigned short* __restrict__ outB) {
  __shared__ float red[256];
  const int r = blockIdx.x, t = threadIdx.x;
  const float* pa = ra + (long)r * 512;
  const float* pb = rb + (long)r * 512;
  float x0 = pa[t] + pb[t];
  float x1 = pa[t + 256] + pb[t + 256];

  red[t] = x0 + x1;
  __syncthreads();
  for (int s = 128; s > 0; s >>= 1) {
    if (t < s) red[t] += red[t + s];
    __syncthreads();
  }
  const float mean = red[0] * (1.f / 512.f);
  __syncthreads();

  const float d0 = x0 - mean, d1 = x1 - mean;
  red[t] = d0 * d0 + d1 * d1;
  __syncthreads();
  for (int s = 128; s > 0; s >>= 1) {
    if (t < s) red[t] += red[t + s];
    __syncthreads();
  }
  const float inv = rsqrtf(red[0] * (1.f / 512.f) + LN_EPS);

  const float y0 = d0 * inv * g[t] + be[t];
  const float y1 = d1 * inv * g[t + 256] + be[t + 256];
  const long o = (long)r * 512;
  outF[o + t] = y0;
  outF[o + t + 256] = y1;
  if (outB) {
    outB[o + t] = f2bf(y0);
    outB[o + t + 256] = f2bf(y1);
  }
}

// ---------------------------------------------------------------------------
extern "C" void kernel_launch(void* const* d_in, const int* in_sizes, int n_in,
                              void* d_out, int out_size, void* d_ws, size_t ws_size,
                              hipStream_t stream) {
  (void)in_sizes; (void)n_in; (void)out_size; (void)ws_size;
  const float* q   = (const float*)d_in[0];
  const float* k   = (const float*)d_in[1];
  const float* v   = (const float*)d_in[2];
  const float* Wq  = (const float*)d_in[3];
  const float* bq  = (const float*)d_in[4];
  const float* Wk  = (const float*)d_in[5];
  const float* bk  = (const float*)d_in[6];
  const float* Wv  = (const float*)d_in[7];
  const float* bv  = (const float*)d_in[8];
  const float* Wo  = (const float*)d_in[9];
  const float* bo  = (const float*)d_in[10];
  const float* tkp = (const float*)d_in[11];
  const float* A1  = (const float*)d_in[12];
  const float* B1  = (const float*)d_in[13];
  const float* A2  = (const float*)d_in[14];
  const float* B2  = (const float*)d_in[15];
  const float* g1  = (const float*)d_in[16];
  const float* be1 = (const float*)d_in[17];
  const float* g2  = (const float*)d_in[18];
  const float* be2 = (const float*)d_in[19];

  // ---- workspace carve-up (256B aligned) ----
  char* ws = (char*)d_ws;
  size_t off = 0;
  auto alloc = [&](size_t bytes) -> void* {
    void* p = ws + off;
    off += (bytes + 255) & ~(size_t)255;
    return p;
  };
  const size_t ACT = (size_t)M_ROWS * 512;          // 4,194,304 elements
  float*          scores = (float*)alloc(64ull * 1024 * 1024 * 4); // 256 MB
  unsigned short* qbf  = (unsigned short*)alloc(ACT * 2);
  unsigned short* kbf  = (unsigned short*)alloc(ACT * 2);
  unsigned short* vbf  = (unsigned short*)alloc(ACT * 2);
  unsigned short* Qb   = (unsigned short*)alloc(ACT * 2);
  unsigned short* Kb   = (unsigned short*)alloc(ACT * 2);
  unsigned short* Vb   = (unsigned short*)alloc(ACT * 2);
  unsigned short* VT   = (unsigned short*)alloc(ACT * 2);
  unsigned short* WqT  = (unsigned short*)alloc(512 * 512 * 2);
  unsigned short* WkT  = (unsigned short*)alloc(512 * 512 * 2);
  unsigned short* WvT  = (unsigned short*)alloc(512 * 512 * 2);
  unsigned short* WoT  = (unsigned short*)alloc(512 * 512 * 2);
  unsigned short* A1T  = (unsigned short*)alloc(512 * 64 * 2);
  unsigned short* B1T  = (unsigned short*)alloc(64 * 2048 * 2);
  unsigned short* A2T  = (unsigned short*)alloc(2048 * 64 * 2);
  unsigned short* B2T  = (unsigned short*)alloc(64 * 512 * 2);
  unsigned short* AO   = (unsigned short*)alloc(ACT * 2);
  float*          attn = (float*)alloc(ACT * 4);
  float*          src  = (float*)alloc(ACT * 4);
  unsigned short* srcb = (unsigned short*)alloc(ACT * 2);
  unsigned short* t1   = (unsigned short*)alloc((size_t)M_ROWS * 64 * 2);
  unsigned short* t2   = (unsigned short*)alloc((size_t)M_ROWS * 2048 * 2);
  unsigned short* t3   = (unsigned short*)alloc((size_t)M_ROWS * 64 * 2);
  float*          ffn  = (float*)alloc(ACT * 4);

  // ---- 1. convert activations + pack weights as bf16 (W transposed) ----
  cvt_bf16<<<(int)(ACT / 256), 256, 0, stream>>>(q, qbf, (long)ACT);
  cvt_bf16<<<(int)(ACT / 256), 256, 0, stream>>>(k, kbf, (long)ACT);
  cvt_bf16<<<(int)(ACT / 256), 256, 0, stream>>>(v, vbf, (long)ACT);
  transpose_bf16<<<1024, 256, 0, stream>>>(Wq, WqT, 512, 512);
  transpose_bf16<<<1024, 256, 0, stream>>>(Wk, WkT, 512, 512);
  transpose_bf16<<<1024, 256, 0, stream>>>(Wv, WvT, 512, 512);
  transpose_bf16<<<1024, 256, 0, stream>>>(Wo, WoT, 512, 512);
  transpose_bf16<<<128, 256, 0, stream>>>(A1, A1T, 512, 64);
  transpose_bf16<<<512, 256, 0, stream>>>(B1, B1T, 64, 2048);
  transpose_bf16<<<512, 256, 0, stream>>>(A2, A2T, 2048, 64);
  transpose_bf16<<<128, 256, 0, stream>>>(B2, B2T, 64, 512);

  // ---- 2. Q/K/V projections (8192x512x512, bias, bf16 out) ----
  dim3 gproj(M_ROWS / 128, D_MODEL / 64, 1);
  gemm_wmma_bf16<true, false, true><<<gproj, 256, 0, stream>>>(
      qbf, 0, 0, 512, WqT, 0, 0, 512, Qb, 0, 0, 512, 1, bq, 1.f, 512);
  gemm_wmma_bf16<true, false, true><<<gproj, 256, 0, stream>>>(
      kbf, 0, 0, 512, WkT, 0, 0, 512, Kb, 0, 0, 512, 1, bk, 1.f, 512);
  gemm_wmma_bf16<true, false, true><<<gproj, 256, 0, stream>>>(
      vbf, 0, 0, 512, WvT, 0, 0, 512, Vb, 0, 0, 512, 1, bv, 1.f, 512);
  transpose_v<<<(int)(ACT / 256), 256, 0, stream>>>(Vb, VT);

  // ---- 3. scores = Qh @ Kh^T / sqrt(64), per (n,h) batch (z = n*8+h) ----
  dim3 gsc(SEQ_L / 128, SEQ_L / 64, 64);
  gemm_wmma_bf16<false, false, false><<<gsc, 256, 0, stream>>>(
      Qb, 512, 64, 4096,               // A base = 512*n + 64*h, row stride 4096
      Kb, 512, 64, 4096,               // Bt same indexing (Kh rows = s)
      scores, 8l * 1048576, 1048576, 1024, 1,
      nullptr, 0.125f, 64);

  // ---- 4. exact top-k threshold + masked softmax (probs overlay scores) ----
  topk_softmax<<<64 * SEQ_L, 256, 0, stream>>>(scores, tkp);

  // ---- 5. ao = probs @ Vh, written interleaved as AO[l,n, d*8+h] ----
  dim3 gat(SEQ_L / 128, 1, 64);
  gemm_wmma_bf16<true, false, false><<<gat, 256, 0, stream>>>(
      (const unsigned short*)scores, 8l * 2097152, 2097152, 2048,
      VT, 524288, 65536, 1024,
      AO, 512, 1, 4096, 8,             // C(l,d) -> AO[(l*8+n)*512 + d*8 + h]
      nullptr, 1.f, 1024);

  // ---- 6. attn_out = AO @ Wo + bo (f32) ----
  gemm_wmma_bf16<false, false, true><<<gproj, 256, 0, stream>>>(
      AO, 0, 0, 512, WoT, 0, 0, 512, attn, 0, 0, 512, 1, bo, 1.f, 512);

  // ---- 7. src = LN(q + attn_out) ----
  ln_residual<<<M_ROWS, 256, 0, stream>>>(q, attn, g1, be1, src, srcb);

  // ---- 8. low-rank FFN: relu(src@A1@B1)@A2@B2 ----
  dim3 gr64(M_ROWS / 128, 1, 1);
  gemm_wmma_bf16<true, false, false><<<gr64, 256, 0, stream>>>(
      srcb, 0, 0, 512, A1T, 0, 0, 512, t1, 0, 0, 64, 1, nullptr, 1.f, 512);
  dim3 gff(M_ROWS / 128, DFF / 64, 1);
  gemm_wmma_bf16<true, true, false><<<gff, 256, 0, stream>>>(
      t1, 0, 0, 64, B1T, 0, 0, 64, t2, 0, 0, 2048, 1, nullptr, 1.f, 64);
  gemm_wmma_bf16<true, false, false><<<gr64, 256, 0, stream>>>(
      t2, 0, 0, 2048, A2T, 0, 0, 2048, t3, 0, 0, 64, 1, nullptr, 1.f, 2048);
  dim3 gout(M_ROWS / 128, D_MODEL / 64, 1);
  gemm_wmma_bf16<false, false, false><<<gout, 256, 0, stream>>>(
      t3, 0, 0, 64, B2T, 0, 0, 64, ffn, 0, 0, 512, 1, nullptr, 1.f, 64);

  // ---- 9. out = LN(src + ffn) ----
  ln_residual<<<M_ROWS, 256, 0, stream>>>(src, ffn, g2, be2, (float*)d_out,
                                          nullptr);
}
